// BoTMultiHeadAttention_30236569764577
// MI455X (gfx1250) — compile-verified, matrix operand order checked by reading, not run on hardware
//
#include <hip/hip_runtime.h>
#include <hip/hip_bf16.h>

// ---------------------------------------------------------------------------
// BoT Multi-Head Attention block for MI455X (gfx1250), BF16 WMMA pipeline.
//   B=2, N=2048, D=1024, H=16, dh=64.
//   att = softmax( (q k^T + q p^T) * s ) = softmax( q (k+p)^T * s )
// All WMMA fragments are loaded as two 16-byte runs per lane (b128 path).
// ---------------------------------------------------------------------------

typedef __bf16 bf16;
typedef __attribute__((ext_vector_type(16))) __bf16 v16bf;
typedef __attribute__((ext_vector_type(8)))  __bf16 v8bf;
typedef __attribute__((ext_vector_type(8)))  float  v8f;

#define BATCH 2
#define NSEQ  2048
#define DMODEL 1024
#define HEADS 16
#define DH    64
#define BN    (BATCH * NSEQ)   // 4096 rows
#define LN_EPS 1e-5f

__device__ __forceinline__ v8f vzero8() {
  v8f z = {0.f, 0.f, 0.f, 0.f, 0.f, 0.f, 0.f, 0.f};
  return z;
}

// CDNA5 16-bit A/B fragment (ISA 7.12.2): lane group khalf = 8*(lane>=16).
// Fragment element order per lane is two contiguous K runs:
//   [khalf, khalf+8) then [16+khalf, 16+khalf+8)
// => two aligned 16-byte loads + register concat.
__device__ __forceinline__ v16bf load_frag(const bf16* p, int khalf) {
  v8bf lo = *(const v8bf*)(p + khalf);
  v8bf hi = *(const v8bf*)(p + 16 + khalf);
  return __builtin_shufflevector(lo, hi, 0, 1, 2, 3, 4, 5, 6, 7, 8, 9, 10, 11,
                                 12, 13, 14, 15);
}

__device__ __forceinline__ v8f wmma_bf16(v16bf a, v16bf b, v8f c) {
  // D = A(16x32 bf16) * B(32x16 bf16) + C(16x16 f32)
  return __builtin_amdgcn_wmma_f32_16x16x32_bf16(
      /*neg_a=*/false, a, /*neg_b=*/false, b,
      /*c_mod=*/(short)0, c, /*reuse_a=*/false, /*reuse_b=*/false);
}

// ---------------------------------------------------------------------------
// Kernel 1: fp32 -> bf16 packing (vectorized: float4 x2 in, v8bf out).
// ---------------------------------------------------------------------------
__global__ __launch_bounds__(256) void pack_kernel(
    const float* __restrict__ qs, const float* __restrict__ Wq,
    const float* __restrict__ Wk, const float* __restrict__ Wv,
    const float* __restrict__ Wo, bf16* __restrict__ X16,
    bf16* __restrict__ W16 /* [4][D*D] : q,k,v,o */) {
  const size_t nx = (size_t)BN * DMODEL;      // 2^22, multiple of 8
  const size_t nw = (size_t)DMODEL * DMODEL;  // 2^20
  const size_t nchunks = (nx + 4 * nw) / 8;
  for (size_t c = (size_t)blockIdx.x * blockDim.x + threadIdx.x; c < nchunks;
       c += (size_t)gridDim.x * blockDim.x) {
    size_t base = c * 8;
    const float* src;
    bf16* dst;
    if (base < nx) {
      src = qs + base;
      dst = X16 + base;
    } else {
      size_t j = base - nx;
      int m = (int)(j >> 20);
      size_t r = j & (nw - 1);
      const float* w = (m == 0) ? Wq : (m == 1) ? Wk : (m == 2) ? Wv : Wo;
      src = w + r;
      dst = W16 + (size_t)m * nw + r;
    }
    float4 a = *(const float4*)(src);
    float4 b = *(const float4*)(src + 4);
    v8bf o;
    o[0] = (bf16)a.x; o[1] = (bf16)a.y; o[2] = (bf16)a.z; o[3] = (bf16)a.w;
    o[4] = (bf16)b.x; o[5] = (bf16)b.y; o[6] = (bf16)b.z; o[7] = (bf16)b.w;
    *(v8bf*)dst = o;
  }
}

// ---------------------------------------------------------------------------
// Kernel 2: fused QKV projection; weights staged TRANSPOSED [col][k] in LDS
// so B-fragments are two ds_load_b128 per WMMA.  Writes Q, KP(=K+pos), V in
// [B,H,N,64] bf16.  Block: 128 rows x 1 head; 8 waves x 4 col-tiles.
// ---------------------------------------------------------------------------
__global__ __launch_bounds__(256) void qkv_kernel(
    const bf16* __restrict__ X16, const bf16* __restrict__ W16,
    const float* __restrict__ pos_emb, bf16* __restrict__ Q16,
    bf16* __restrict__ KP16, bf16* __restrict__ V16) {
  // [matrix][col 0..63][k 0..31, padded 40] ; stride 80B (16B-mult, 64-bank ok)
  __shared__ alignas(16) bf16 ldsW[3][DH][40];

  const int lane = threadIdx.x & 31;
  const int wave = threadIdx.x >> 5;
  const int head = blockIdx.y;
  const int wrow = blockIdx.x * 128 + wave * 16;
  const int col0 = head * DH;
  const int arow = wrow + (lane & 15);
  const int khalf = (lane & 16) ? 8 : 0;

  v8f acc[3][4];
#pragma unroll
  for (int m = 0; m < 3; ++m)
#pragma unroll
    for (int c = 0; c < 4; ++c) acc[m][c] = vzero8();

  for (int k0 = 0; k0 < DMODEL; k0 += 32) {
    // Stage 32x64 slices of Wq,Wk,Wv transposed: b128 global load, 8 b16
    // scattered LDS stores per thread-iteration (3 iterations).
    for (int e = threadIdx.x; e < 3 * 32 * 8; e += 256) {
      int m = e >> 8;              // matrix
      int rem = e & 255;
      int kk = rem >> 3;           // k row 0..31
      int c8 = (rem & 7) << 3;     // col chunk base
      v8bf chunk = *(const v8bf*)&W16[(size_t)m * DMODEL * DMODEL +
                                      (size_t)(k0 + kk) * DMODEL + col0 + c8];
#pragma unroll
      for (int j = 0; j < 8; ++j) ldsW[m][c8 + j][kk] = chunk[j];
    }
    __syncthreads();

    // A fragment: 16x32 of X (row-major global), reused across 12 WMMAs.
    const bf16* xp = X16 + (size_t)arow * DMODEL + k0;
    if (k0 + 32 < DMODEL) __builtin_prefetch(xp + 32, 0, 1);
    v16bf afrag = load_frag(xp, khalf);

#pragma unroll
    for (int m = 0; m < 3; ++m) {
#pragma unroll
      for (int ct = 0; ct < 4; ++ct) {
        v16bf bfrag = load_frag(&ldsW[m][ct * 16 + (lane & 15)][0], khalf);
        acc[m][ct] = wmma_bf16(afrag, bfrag, acc[m][ct]);
      }
    }
    __syncthreads();
  }

  // Epilogue: scatter C-layout accumulators into [B,H,N,64] bf16 tensors.
  const int hi8 = (lane & 16) ? 8 : 0;
#pragma unroll
  for (int ct = 0; ct < 4; ++ct) {
#pragma unroll
    for (int r = 0; r < 8; ++r) {
      int grow = wrow + r + hi8;  // 0..4095
      int b = grow >> 11;
      int n = grow & (NSEQ - 1);
      int c = ct * 16 + (lane & 15);  // 0..63
      size_t hoff = (((size_t)(b * HEADS + head) * NSEQ) + n) * DH + c;
      Q16[hoff] = (bf16)acc[0][ct][r];
      float pe = pos_emb[(size_t)grow * DMODEL + col0 + c];
      KP16[hoff] = (bf16)(acc[1][ct][r] + pe);  // fold position into key
      V16[hoff] = (bf16)acc[2][ct][r];
    }
  }
}

// ---------------------------------------------------------------------------
// Kernel 3: flash attention per (b,h).  KP staged row-major [key][dh] (dh
// contiguous -> direct b128 fragment reads); V staged transposed [dhcol][key];
// P round-trips through LDS row-major [qrow][k].  Key chunks of 32.
// ---------------------------------------------------------------------------
__global__ __launch_bounds__(256) void attn_kernel(
    const bf16* __restrict__ Q16, const bf16* __restrict__ KP16,
    const bf16* __restrict__ V16, bf16* __restrict__ O16) {
  __shared__ alignas(16) bf16 ldsKP[32][72];     // [key][dh, pad 72] 4.5 KB
  __shared__ alignas(16) bf16 ldsVT[DH][40];     // [dhcol][key, pad 40] 5 KB
  __shared__ alignas(16) bf16 ldsP[8][16][40];   // per-wave P  10 KB

  const int lane = threadIdx.x & 31;
  const int wave = threadIdx.x >> 5;
  const int bh = blockIdx.y;  // 0..31
  const int b = bh >> 4;
  const int h = bh & (HEADS - 1);
  const int n0 = blockIdx.x * 128 + wave * 16;  // query row tile
  const int hi8 = (lane & 16) ? 8 : 0;
  const int khalf = hi8;  // same expression for fragments
  const float scale = 0.125f;  // dh^-0.5

  const bf16* Qh = Q16 + (size_t)bh * NSEQ * DH;
  const bf16* KPh = KP16 + (size_t)bh * NSEQ * DH;
  const bf16* Vh = V16 + (size_t)bh * NSEQ * DH;

  // Q fragments: 16x64 -> two 16x32 A-frags, resident for the whole loop.
  v16bf aq[2];
  {
    const bf16* qp = Qh + (size_t)(n0 + (lane & 15)) * DH;
    aq[0] = load_frag(qp, khalf);
    aq[1] = load_frag(qp + 32, khalf);
  }

  float mrow[8], lrow[8];
#pragma unroll
  for (int r = 0; r < 8; ++r) { mrow[r] = -3.0e38f; lrow[r] = 0.f; }
  v8f oacc[4];
#pragma unroll
  for (int dt = 0; dt < 4; ++dt) oacc[dt] = vzero8();

  for (int kc = 0; kc < NSEQ; kc += 32) {
    // Stage 32 keys (KP row-major) + 32 value rows (transposed).
    for (int e = threadIdx.x; e < 2 * 32 * 8; e += 256) {
      int which = e >> 8;
      int rem = e & 255;
      int kk = rem >> 3;           // key row in chunk
      int c8 = (rem & 7) << 3;     // dh chunk base
      if (which == 0) {
        *(v8bf*)&ldsKP[kk][c8] = *(const v8bf*)&KPh[(size_t)(kc + kk) * DH + c8];
      } else {
        v8bf chunk = *(const v8bf*)&Vh[(size_t)(kc + kk) * DH + c8];
#pragma unroll
        for (int j = 0; j < 8; ++j) ldsVT[c8 + j][kk] = chunk[j];
      }
    }
    __syncthreads();

    // S = Q @ KP^T : 16x32 tile = two 16x16 WMMAs, K = 64 over dh.
    v8f s[2] = {vzero8(), vzero8()};
#pragma unroll
    for (int t = 0; t < 2; ++t) {
#pragma unroll
      for (int ct = 0; ct < 2; ++ct) {
        v16bf bfrag = load_frag(&ldsKP[ct * 16 + (lane & 15)][t * 32], khalf);
        s[ct] = wmma_bf16(aq[t], bfrag, s[ct]);
      }
    }

    // Online softmax.  C-layout: VGPR r holds row (r + hi8), col = lane&15;
    // each row lives in one 16-lane group, so xor masks 1,2,4,8 reduce a row.
    float pv[2][8];
#pragma unroll
    for (int r = 0; r < 8; ++r) {
      float t0 = s[0][r] * scale;
      float t1 = s[1][r] * scale;
      float tm = fmaxf(t0, t1);
#pragma unroll
      for (int m = 1; m < 16; m <<= 1) tm = fmaxf(tm, __shfl_xor(tm, m));
      float nm = fmaxf(mrow[r], tm);
      float alpha = __expf(mrow[r] - nm);
      float p0 = __expf(t0 - nm);
      float p1 = __expf(t1 - nm);
      float ps = p0 + p1;
#pragma unroll
      for (int m = 1; m < 16; m <<= 1) ps += __shfl_xor(ps, m);
      lrow[r] = lrow[r] * alpha + ps;
      mrow[r] = nm;
#pragma unroll
      for (int dt = 0; dt < 4; ++dt) oacc[dt][r] *= alpha;
      pv[0][r] = p0;
      pv[1][r] = p1;
    }

    // P: C-layout -> row-major LDS -> A-fragment (two b128 reads).
#pragma unroll
    for (int ct = 0; ct < 2; ++ct)
#pragma unroll
      for (int r = 0; r < 8; ++r)
        ldsP[wave][r + hi8][ct * 16 + (lane & 15)] = (bf16)pv[ct][r];
    __syncthreads();

    v16bf ap = load_frag(&ldsP[wave][lane & 15][0], khalf);

    // O += P @ V : K = 32 keys, four 16-wide dh tiles, B from ldsVT.
#pragma unroll
    for (int dt = 0; dt < 4; ++dt) {
      v16bf bv = load_frag(&ldsVT[dt * 16 + (lane & 15)][0], khalf);
      oacc[dt] = wmma_bf16(ap, bv, oacc[dt]);
    }
    __syncthreads();
  }

  // Normalize by softmax denominator, write merged-head layout [B,N,D].
#pragma unroll
  for (int dt = 0; dt < 4; ++dt) {
#pragma unroll
    for (int r = 0; r < 8; ++r) {
      int n = n0 + r + hi8;
      float inv = 1.f / lrow[r];
      O16[((size_t)(b * NSEQ + n)) * DMODEL + h * DH + dt * 16 + (lane & 15)] =
          (bf16)(oacc[dt][r] * inv);
    }
  }
}

// ---------------------------------------------------------------------------
// Kernel 4: output projection  Xres = O @ Wo + bo + q_s  (f32 out).
// Wo k-slice staged transposed [col][k]; block 128 rows x 64 cols.
// ---------------------------------------------------------------------------
__global__ __launch_bounds__(256) void oproj_kernel(
    const bf16* __restrict__ O16, const bf16* __restrict__ Wo16,
    const float* __restrict__ bo, const float* __restrict__ qs,
    float* __restrict__ Xres) {
  __shared__ alignas(16) bf16 ldsW[DH][40];  // [col][k pad 40] 5 KB

  const int lane = threadIdx.x & 31;
  const int wave = threadIdx.x >> 5;
  const int wrow = blockIdx.x * 128 + wave * 16;
  const int col0 = blockIdx.y * 64;
  const int arow = wrow + (lane & 15);
  const int khalf = (lane & 16) ? 8 : 0;

  v8f acc[4];
#pragma unroll
  for (int c = 0; c < 4; ++c) acc[c] = vzero8();

  for (int k0 = 0; k0 < DMODEL; k0 += 32) {
    for (int e = threadIdx.x; e < 32 * 8; e += 256) {
      int kk = e >> 3;
      int c8 = (e & 7) << 3;
      v8bf chunk =
          *(const v8bf*)&Wo16[(size_t)(k0 + kk) * DMODEL + col0 + c8];
#pragma unroll
      for (int j = 0; j < 8; ++j) ldsW[c8 + j][kk] = chunk[j];
    }
    __syncthreads();

    const bf16* op = O16 + (size_t)arow * DMODEL + k0;
    if (k0 + 32 < DMODEL) __builtin_prefetch(op + 32, 0, 1);
    v16bf afrag = load_frag(op, khalf);

#pragma unroll
    for (int ct = 0; ct < 4; ++ct) {
      v16bf bfrag = load_frag(&ldsW[ct * 16 + (lane & 15)][0], khalf);
      acc[ct] = wmma_bf16(afrag, bfrag, acc[ct]);
    }
    __syncthreads();
  }

  const int hi8 = (lane & 16) ? 8 : 0;
#pragma unroll
  for (int ct = 0; ct < 4; ++ct) {
#pragma unroll
    for (int r = 0; r < 8; ++r) {
      int grow = wrow + r + hi8;
      int c = col0 + ct * 16 + (lane & 15);
      float v = acc[ct][r] + bo[c] + qs[(size_t)grow * DMODEL + c];
      Xres[(size_t)grow * DMODEL + c] = v;
    }
  }
}

// ---------------------------------------------------------------------------
// Kernel 5: row-wise layernorm, one block per row (float4 path).
// ---------------------------------------------------------------------------
__global__ __launch_bounds__(256) void ln_kernel(
    const float* __restrict__ Xres, const float* __restrict__ gamma,
    const float* __restrict__ beta, float* __restrict__ out) {
  const int row = blockIdx.x;
  const float* x = Xres + (size_t)row * DMODEL;
  __shared__ float red[2][8];

  const int c4 = threadIdx.x * 4;  // 256 threads x 4 = 1024
  float4 xv = *(const float4*)(x + c4);
  float s = xv.x + xv.y + xv.z + xv.w;
  float ss = xv.x * xv.x + xv.y * xv.y + xv.z * xv.z + xv.w * xv.w;
#pragma unroll
  for (int m = 1; m < 32; m <<= 1) {
    s += __shfl_xor(s, m);
    ss += __shfl_xor(ss, m);
  }
  const int wave = threadIdx.x >> 5;
  const int lane = threadIdx.x & 31;
  if (lane == 0) {
    red[0][wave] = s;
    red[1][wave] = ss;
  }
  __syncthreads();
  if (wave == 0) {
    float a = (lane < 8) ? red[0][lane] : 0.f;
    float q = (lane < 8) ? red[1][lane] : 0.f;
#pragma unroll
    for (int m = 1; m < 8; m <<= 1) {
      a += __shfl_xor(a, m);
      q += __shfl_xor(q, m);
    }
    if (lane == 0) {
      red[0][0] = a;
      red[1][0] = q;
    }
  }
  __syncthreads();
  float mu = red[0][0] * (1.f / DMODEL);
  float var = red[1][0] * (1.f / DMODEL) - mu * mu;
  float inv = rsqrtf(var + LN_EPS);
  float4 g = *(const float4*)(gamma + c4);
  float4 bb = *(const float4*)(beta + c4);
  float4 o;
  o.x = (xv.x - mu) * inv * g.x + bb.x;
  o.y = (xv.y - mu) * inv * g.y + bb.y;
  o.z = (xv.z - mu) * inv * g.z + bb.z;
  o.w = (xv.w - mu) * inv * g.w + bb.w;
  *(float4*)(out + (size_t)row * DMODEL + c4) = o;
}

// ---------------------------------------------------------------------------
// Host launcher.
// ---------------------------------------------------------------------------
extern "C" void kernel_launch(void* const* d_in, const int* in_sizes, int n_in,
                              void* d_out, int out_size, void* d_ws,
                              size_t ws_size, hipStream_t stream) {
  (void)in_sizes; (void)n_in; (void)out_size; (void)ws_size;
  const float* qs = (const float*)d_in[0];
  const float* pos = (const float*)d_in[1];
  const float* Wq = (const float*)d_in[2];
  const float* Wk = (const float*)d_in[3];
  const float* Wv = (const float*)d_in[4];
  const float* Wo = (const float*)d_in[5];
  const float* bo = (const float*)d_in[6];
  const float* lng = (const float*)d_in[7];
  const float* lnb = (const float*)d_in[8];

  char* ws = (char*)d_ws;
  size_t off = 0;
  const size_t act_b = (size_t)BN * DMODEL * sizeof(bf16);   // 8 MB
  const size_t w_b = 4ull * DMODEL * DMODEL * sizeof(bf16);  // 8 MB
  bf16* X16 = (bf16*)(ws + off); off += act_b;
  bf16* W16 = (bf16*)(ws + off); off += w_b;
  bf16* Q16 = (bf16*)(ws + off); off += act_b;
  bf16* KP16 = (bf16*)(ws + off); off += act_b;
  bf16* V16 = (bf16*)(ws + off); off += act_b;
  bf16* O16 = (bf16*)(ws + off); off += act_b;
  float* Xres = (float*)(ws + off); off += (size_t)BN * DMODEL * sizeof(float);

  pack_kernel<<<1024, 256, 0, stream>>>(qs, Wq, Wk, Wv, Wo, X16, W16);
  qkv_kernel<<<dim3(BN / 128, HEADS), 256, 0, stream>>>(X16, W16, pos, Q16,
                                                        KP16, V16);
  attn_kernel<<<dim3(NSEQ / 128, BATCH * HEADS), 256, 0, stream>>>(Q16, KP16,
                                                                   V16, O16);
  oproj_kernel<<<dim3(BN / 128, DMODEL / 64), 256, 0, stream>>>(
      O16, W16 + 3ull * DMODEL * DMODEL, bo, qs, Xres);
  ln_kernel<<<BN, 256, 0, stream>>>(Xres, lng, lnb, (float*)d_out);
}
